// DilatedCorrelation_43525198578382
// MI455X (gfx1250) — matrix operand order repaired and checked by reading, not compiled, not run on hardware
//
#include <hip/hip_runtime.h>
#include <hip/hip_bf16.h>

typedef __attribute__((ext_vector_type(16))) _Float16 v16h;
typedef __attribute__((ext_vector_type(8)))  _Float16 v8h;
typedef __attribute__((ext_vector_type(8)))  float    v8f;
typedef __attribute__((ext_vector_type(4)))  int      v4i;

#define B_  8
#define C_  256
#define H_  64
#define W_  128
#define KOF 243          // 3 dilations * 9 * 9
#define ASTRIDE 264      // padded fp16 row stride (528B, bank-friendly)
#define BCOLS 160        // 128 cols + 16 halo each side

#if __has_builtin(__builtin_amdgcn_global_load_async_to_lds_b128)
#define USE_ASYNC 1
typedef __attribute__((address_space(1))) v4i gv4i;   // global 16B chunk
typedef __attribute__((address_space(3))) v4i lv4i;   // LDS 16B chunk
#endif

// ---------------------------------------------------------------------------
// Kernel 1: L2-normalize over channels, convert fp32 NCHW -> fp16 NHWC.
// ---------------------------------------------------------------------------
__global__ __launch_bounds__(256) void nrm_kernel(const float* __restrict__ src,
                                                  _Float16* __restrict__ dst) {
  __shared__ float tile[32][258];
  __shared__ float part[8][32];
  __shared__ float inv[32];
  const int tid = threadIdx.x;
  const int tx = tid & 31, ty = tid >> 5;
  const int w0 = blockIdx.x * 32, h = blockIdx.y, b = blockIdx.z;
  const size_t pix = (size_t)h * W_ + w0 + tx;

  float s = 0.f;
  for (int c = ty; c < C_; c += 8) {
    float v = src[((size_t)b * C_ + c) * ((size_t)H_ * W_) + pix];
    tile[tx][c] = v;
    s += v * v;
  }
  part[ty][tx] = s;
  __syncthreads();
  if (tid < 32) {
    float t = 0.f;
    for (int j = 0; j < 8; ++j) t += part[j][tid];
    inv[tid] = 1.0f / fmaxf(sqrtf(t), 1e-12f);   // matches F.normalize eps
  }
  __syncthreads();

  uint4* dptr = (uint4*)(dst + (((size_t)b * H_ + h) * W_ + w0) * C_);
  for (int q = tid; q < 1024; q += 256) {
    int p = q >> 5, c0 = (q & 31) * 8;
    float iv = inv[p];
    union { uint4 u; _Float16 hh[8]; } pk;
#pragma unroll
    for (int t = 0; t < 8; ++t) pk.hh[t] = (_Float16)(tile[p][c0 + t] * iv);
    dptr[q] = pk.u;
  }
}

// ---------------------------------------------------------------------------
// Staging: 128 rows x 256ch fp16 (64KB) -> LDS with padded row stride.
// Async DMA to LDS when available (ASYNCcnt), else register copy.
// ---------------------------------------------------------------------------
__device__ __forceinline__ void stage_row(_Float16* ldsBase,
                                          const _Float16* g, int tid) {
#pragma unroll
  for (int t = 0; t < 8; ++t) {
    int i = tid + t * 512;
    int r = i >> 5, o = i & 31;
    _Float16* lp = ldsBase + r * ASTRIDE + o * 8;
    const _Float16* gp = g + (size_t)i * 8;
#ifdef USE_ASYNC
    __builtin_amdgcn_global_load_async_to_lds_b128((gv4i*)gp, (lv4i*)lp, 0, 0);
#else
    *(uint4*)lp = *(const uint4*)gp;
#endif
  }
}

__device__ __forceinline__ void wait_staging() {
#ifdef USE_ASYNC
#if __has_builtin(__builtin_amdgcn_s_wait_asynccnt)
  __builtin_amdgcn_s_wait_asynccnt(0);
#else
  asm volatile("s_wait_asynccnt 0x0" ::: "memory");
#endif
#endif
}

// ---------------------------------------------------------------------------
// Kernel 2: WMMA correlation, one block per (b,h) row, 16 waves.
// Double-buffered async f2-row staging overlapped with compute.
// ---------------------------------------------------------------------------
union Frag { v16h v; v8h p[2]; };

__device__ __forceinline__ void load_a(Frag& f, const _Float16* Ap, int off) {
  f.p[0] = *(const v8h*)(Ap + off);
  f.p[1] = *(const v8h*)(Ap + off + 16);
}
__device__ __forceinline__ void load_b(Frag& f, const _Float16* Bp, int off) {
  f.p[0] = *(const v8h*)(Bp + off);
  f.p[1] = *(const v8h*)(Bp + off + 16);
}

__global__ __launch_bounds__(512) void corr_kernel(const _Float16* __restrict__ f1n,
                                                   const _Float16* __restrict__ f2n,
                                                   float* __restrict__ out) {
  extern __shared__ _Float16 smem[];
  _Float16* Af = smem;                               // 128 x ASTRIDE
  _Float16* Bbuf0 = smem + 128 * ASTRIDE;            // 160 x ASTRIDE (ping)
  _Float16* Bbuf1 = Bbuf0 + BCOLS * ASTRIDE;         // 160 x ASTRIDE (pong)

  const int tid    = threadIdx.x;
  const int wave   = tid >> 5;
  const int lane   = tid & 31;
  const int laneHi = lane >> 4;
  const int mn     = lane & 15;
  const int h = blockIdx.x, b = blockIdx.y;
  const _Float16* f2base = f2n + ((size_t)b * H_) * W_ * C_;

  // One-time staging: f1 row + zero halo columns of both B buffers.
  stage_row(Af, f1n + ((size_t)b * H_ + h) * W_ * C_, tid);
  {
    uint4 z; z.x = z.y = z.z = z.w = 0u;
    for (int i = tid; i < 2048; i += 512) {          // 2 bufs * 32 halo cols * 32
      int buf = i >> 10, j = i & 1023;
      int p = j >> 5, o = j & 31;
      int col = (p < 16) ? p : (p + 128);
      _Float16* Bb = buf ? Bbuf1 : Bbuf0;
      ((uint4*)(Bb + col * ASTRIDE))[o] = z;
    }
  }
  // Prologue: stage row for iteration 0 (if in bounds).
  {
    int h2 = h + (-4) * 1;                            // it=0: d=1, dy=-4
    if (h2 >= 0 && h2 < H_)
      stage_row(Bbuf0 + 16 * ASTRIDE, f2base + (size_t)h2 * W_ * C_, tid);
  }

  for (int it = 0; it < 27; ++it) {
    wait_staging();          // this wave's DMA (row `it` + first-time A) done
    __syncthreads();         // all waves' staging visible; prev compute done

    // Issue DMA for row it+1 into the other buffer (overlaps with compute).
    if (it + 1 < 27) {
      int diN = (it + 1) / 9;
      int hn = h + ((it + 1) % 9 - 4) * (1 << diN);
      if (hn >= 0 && hn < H_) {
        _Float16* Bn = ((it + 1) & 1) ? Bbuf1 : Bbuf0;
        stage_row(Bn + 16 * ASTRIDE, f2base + (size_t)hn * W_ * C_, tid);
      }
    }

    const int dIdx = it / 9;
    const int d = 1 << dIdx;                          // 1,2,4
    const int dy = it % 9 - 4;
    const int kbase = dIdx * 81 + (dy + 4) * 9;
    const int h2 = h + dy * d;

    if (h2 >= 0 && h2 < H_) {
      const _Float16* Bf = (it & 1) ? Bbuf1 : Bbuf0;
      const int ntiles = (dIdx == 2) ? 3 : 2;
      const int njobs = 8 * ntiles;
      for (int job = wave; job < njobs; job += 16) {  // wave-uniform
        const int at = job / ntiles;
        const int nt = job - at * ntiles;
        const int w0 = at * 16;
        const int colbase = w0 - 4 * d + nt * 16;     // abs w of B col 0

        const _Float16* Ap = Af + (w0 + mn) * ASTRIDE + laneHi * 8;
        const _Float16* Bp = Bf + (colbase + 16 + mn) * ASTRIDE + laneHi * 8;

        // Software-pipelined K loop: fragments for step i+1 load while
        // WMMA i executes (waits become dscnt<=4 instead of 0).
        Frag fa0, fb0, fa1, fb1;
        load_a(fa0, Ap, 0);
        load_b(fb0, Bp, 0);
        v8f acc = {};
#pragma unroll
        for (int i = 0; i < 8; ++i) {
          if ((i & 1) == 0) {
            if (i < 7) { load_a(fa1, Ap, (i + 1) * 32); load_b(fb1, Bp, (i + 1) * 32); }
            acc = __builtin_amdgcn_wmma_f32_16x16x32_f16(
                false, fa0.v, false, fb0.v, (short)0, acc, false, false);
          } else {
            if (i < 7) { load_a(fa0, Ap, (i + 1) * 32); load_b(fb0, Bp, (i + 1) * 32); }
            acc = __builtin_amdgcn_wmma_f32_16x16x32_f16(
                false, fa1.v, false, fb1.v, (short)0, acc, false, false);
          }
        }

        // Diagonal extraction: C[M,N] -> out(pixel w0+M, dx) iff N-M == dx*d.
        const int col = colbase + mn;
#pragma unroll
        for (int j = 0; j < 8; ++j) {
          const int wp = w0 + j + 8 * laneHi;
          const int delta = col - wp;
          if ((delta & (d - 1)) == 0) {
            const int dx = delta >> dIdx;
            if (dx >= -4 && dx <= 4) {
              const int k = kbase + dx + 4;
              out[(((size_t)b * KOF + k) * H_ + h) * W_ + wp] = acc[j];
            }
          }
        }
      }
    } else {
      // whole shifted row is zero padding -> these 9 k-slices are zero
      for (int i = tid; i < 9 * W_; i += 512) {
        int k = kbase + i / W_;
        out[(((size_t)b * KOF + k) * H_ + h) * W_ + (i % W_)] = 0.f;
      }
    }
  }
}

// ---------------------------------------------------------------------------
extern "C" void kernel_launch(void* const* d_in, const int* in_sizes, int n_in,
                              void* d_out, int out_size, void* d_ws, size_t ws_size,
                              hipStream_t stream) {
  const float* f1 = (const float*)d_in[0];
  const float* f2 = (const float*)d_in[1];
  float* out = (float*)d_out;

  _Float16* f1n = (_Float16*)d_ws;                 // 32 MiB
  _Float16* f2n = f1n + (size_t)B_ * H_ * W_ * C_; // 32 MiB (needs >=64 MiB ws)

  dim3 g1(W_ / 32, H_, B_);
  nrm_kernel<<<g1, 256, 0, stream>>>(f1, f1n);
  nrm_kernel<<<g1, 256, 0, stream>>>(f2, f2n);

  const int smemBytes =
      (128 * ASTRIDE + 2 * BCOLS * ASTRIDE) * (int)sizeof(_Float16); // 236,544 B
  (void)hipFuncSetAttribute((const void*)corr_kernel,
                            hipFuncAttributeMaxDynamicSharedMemorySize, smemBytes);
  corr_kernel<<<dim3(H_, B_), 512, smemBytes, stream>>>(f1n, f2n, out);
}
